// Model_39676907884578
// MI455X (gfx1250) — matrix operand order repaired and checked by reading, not compiled, not run on hardware
//
#include <hip/hip_runtime.h>

typedef __attribute__((ext_vector_type(16))) __bf16 v16bf;
typedef __attribute__((ext_vector_type(8)))  float  v8f;

namespace {
constexpr int Bc = 2, Hc = 16, Sc = 2048, Dc = 64;
constexpr int KT    = 32;          // keys per main-loop step
constexpr int WAVES = 4;           // waves per block (wave32)
constexpr int KSTR = Dc + 4;       // padded LDS row (bf16 elems) for K tile [key][d]
constexpr int VSTR = KT + 2;       // padded LDS row for V tile, d-major [d][key]
constexpr int PSTR = KT + 2;       // padded LDS row for P tile [q][key]
// fold 1/sqrt(D) and log2(e) into Q so that exp(s/8) == exp2(score)
constexpr float QSCALE = 0.125f * 1.44269504088896340736f;
}

// K index of element e in a 16-bit A-matrix (16x32) v16bf operand, per ISA table:
// lanes 0-15:  V0..3 -> K=0..7,  V4..7 -> K=16..23
// lanes 16-31: V0..3 -> K=8..15, V4..7 -> K=24..31
__device__ __forceinline__ int kA(int e, int lane) {
  return ((e >> 3) << 4) + (((lane >> 4) & 1) << 3) + (((e >> 1) & 3) << 1) + (e & 1);
}

__global__ __launch_bounds__(WAVES * 32) void
fattn(const float* __restrict__ Qg, const float* __restrict__ Kg,
      const float* __restrict__ Vg, const float* __restrict__ Mg,
      float* __restrict__ Og) {
  __shared__ __bf16 Kb[KT * KSTR];        // [key][d]
  __shared__ __bf16 Vt[Dc * VSTR];        // [d][key]  (transposed for B gathers)
  __shared__ __bf16 Pt[WAVES][16 * PSTR]; // [q][key] per wave

  const int tid  = threadIdx.x;
  const int lane = tid & 31;
  const int wave = tid >> 5;
  const int col  = lane & 15;        // N (column) within 16-wide tiles
  const int hi   = (lane >> 4) & 1;  // 0: lanes 0-15, 1: lanes 16-31

  const int qTiles = Sc / (WAVES * 16);           // 32 q-tiles per head
  const int bh = blockIdx.x / qTiles;
  const int qt = blockIdx.x % qTiles;
  const int q0 = qt * (WAVES * 16) + wave * 16;

  const size_t headQ = (size_t)bh * Sc * Dc;
  const size_t headM = (size_t)bh * Sc * Sc;

  const v8f vzero = {0.f, 0.f, 0.f, 0.f, 0.f, 0.f, 0.f, 0.f};

  // ---- load Q tile (16x64) once into two A-layout bf16 register tiles,
  //      pre-scaled by 1/sqrt(D)*log2(e) so scores feed exp2 directly ----
  v16bf qa0, qa1;
  {
    const float* qrow = Qg + headQ + (size_t)(q0 + col) * Dc;  // A: M = lane%16
#pragma unroll
    for (int e = 0; e < 16; ++e) {
      const int k = hi * 16 + e;
      qa0[e] = (__bf16)(qrow[k] * QSCALE);
      qa1[e] = (__bf16)(qrow[32 + k] * QSCALE);
    }
  }

  // ---- O accumulator (16x64 f32 = 4 C-tiles) + per-lane partial denominators ----
  v8f acc[4];
#pragma unroll
  for (int t = 0; t < 4; ++t) acc[t] = vzero;
  float rsum[8];
#pragma unroll
  for (int r = 0; r < 8; ++r) rsum[r] = 0.0f;

  for (int k0 = 0; k0 < Sc; k0 += KT) {
    __syncthreads();
    // cooperative stage of K/V 32x64 f32 tiles (contiguous in memory) -> bf16 LDS
    {
      const float* ksrc = Kg + headQ + (size_t)k0 * Dc;
      const float* vsrc = Vg + headQ + (size_t)k0 * Dc;
#pragma unroll
      for (int j = 0; j < 4; ++j) {
        const int idx = j * 512 + tid * 4;                 // 0..2047, float index
        const int row = idx >> 6, cidx = idx & 63;         // row = key, cidx = d
        const float4 kf = *(const float4*)(ksrc + idx);
        const float4 vf = *(const float4*)(vsrc + idx);
        __bf16* kd = &Kb[row * KSTR + cidx];
        kd[0] = (__bf16)kf.x; kd[1] = (__bf16)kf.y;
        kd[2] = (__bf16)kf.z; kd[3] = (__bf16)kf.w;
        Vt[(cidx + 0) * VSTR + row] = (__bf16)vf.x;        // transpose: [d][key]
        Vt[(cidx + 1) * VSTR + row] = (__bf16)vf.y;
        Vt[(cidx + 2) * VSTR + row] = (__bf16)vf.z;
        Vt[(cidx + 3) * VSTR + row] = (__bf16)vf.w;
      }
    }
    __syncthreads();

    // ---- scores (pre-scaled): 16 queries x 32 keys (4 WMMAs) ----
    v8f st[2];
#pragma unroll
    for (int kh = 0; kh < 2; ++kh) {
      v16bf b0, b1;  // B-matrix (32x16): lane = N(key), elem e = K(d) + 16*hi
#pragma unroll
      for (int e = 0; e < 16; ++e) {
        const int key = kh * 16 + col;
        const int d   = hi * 16 + e;
        b0[e] = Kb[key * KSTR + d];
        b1[e] = Kb[key * KSTR + 32 + d];
      }
      v8f c = vzero;
      c = __builtin_amdgcn_wmma_f32_16x16x32_bf16(false, qa0, false, b0,
                                                  (short)0, c, false, false);
      c = __builtin_amdgcn_wmma_f32_16x16x32_bf16(false, qa1, false, b1,
                                                  (short)0, c, false, false);
      st[kh] = c;
    }

    // ---- un-shifted softmax numerators via raw v_exp_f32 (scores are O(1),
    //      far from subnormal guard range, so no OCML denorm fixup needed);
    //      mask applied to P only (not denominator); denominator kept as
    //      per-lane partials, reduced once after the loop ----
#pragma unroll
    for (int r = 0; r < 8; ++r) {
      const float p0 = __builtin_amdgcn_exp2f(st[0][r]);
      const float p1 = __builtin_amdgcn_exp2f(st[1][r]);
      rsum[r] += p0 + p1;
      const int m = r + hi * 8;
      const float* mrow = Mg + headM + (size_t)(q0 + m) * Sc + k0;
      Pt[wave][m * PSTR + col]      = (__bf16)(p0 * mrow[col]);
      Pt[wave][m * PSTR + 16 + col] = (__bf16)(p1 * mrow[16 + col]);
    }
    asm volatile("s_wait_dscnt 0" ::: "memory");  // per-wave LDS RAW fence

    // ---- O += P V (re-layout P C->A via LDS, then 4 WMMAs over D) ----
    v16bf pa;
#pragma unroll
    for (int e = 0; e < 16; ++e) pa[e] = Pt[wave][col * PSTR + hi * 16 + e];
#pragma unroll
    for (int dg = 0; dg < 4; ++dg) {
      v16bf vb;  // B-matrix: lane = N(d col), elem e = K(key) + 16*hi; contiguous
#pragma unroll
      for (int e = 0; e < 16; ++e)
        vb[e] = Vt[(dg * 16 + col) * VSTR + hi * 16 + e];
      acc[dg] = __builtin_amdgcn_wmma_f32_16x16x32_bf16(false, pa, false, vb,
                                                        (short)0, acc[dg],
                                                        false, false);
    }
  }

  // ---- one-time denominator reduction across the 16 lanes of each half ----
  float rinv[8];
#pragma unroll
  for (int r = 0; r < 8; ++r) {
    float s = rsum[r];
    s += __shfl_xor(s, 1, 32);
    s += __shfl_xor(s, 2, 32);
    s += __shfl_xor(s, 4, 32);
    s += __shfl_xor(s, 8, 32);
    rinv[r] = __builtin_amdgcn_rcpf(s);
  }

  // ---- normalize and store fp32 output ----
#pragma unroll
  for (int dg = 0; dg < 4; ++dg)
#pragma unroll
    for (int r = 0; r < 8; ++r) {
      const int m = r + hi * 8;
      Og[headQ + (size_t)(q0 + m) * Dc + dg * 16 + col] = acc[dg][r] * rinv[r];
    }
}

extern "C" void kernel_launch(void* const* d_in, const int* in_sizes, int n_in,
                              void* d_out, int out_size, void* d_ws, size_t ws_size,
                              hipStream_t stream) {
  const float* Q = (const float*)d_in[0];
  const float* K = (const float*)d_in[1];
  const float* V = (const float*)d_in[2];
  const float* M = (const float*)d_in[3];
  float* O = (float*)d_out;
  const int blocks = Bc * Hc * (Sc / (WAVES * 16));  // 1024
  fattn<<<blocks, WAVES * 32, 0, stream>>>(Q, K, V, M, O);
}
